// AttentionFusion_49039936586285
// MI455X (gfx1250) — matrix operand order repaired and checked by reading, not compiled, not run on hardware
//
#include <hip/hip_runtime.h>
#include <hip/hip_bf16.h>
#include <math.h>

#define B_    8
#define N_    16384
#define NC_   64
#define C_    256
#define MTILE 128

typedef __attribute__((ext_vector_type(16))) __bf16 v16bf;
typedef __attribute__((ext_vector_type(8)))  __bf16 v8bf;
typedef __attribute__((ext_vector_type(8)))  float  v8f;

__device__ __forceinline__ v16bf combine8(v8bf lo, v8bf hi) {
    v16bf r;
#pragma unroll
    for (int i = 0; i < 8; ++i) { r[i] = lo[i]; r[i + 8] = hi[i]; }
    return r;
}

// ---------------- prep kernels (tiny, feed L2-resident bf16 operands) -------

// WT[n][d] = bf16(W_p[d][n])   (256x256)
__global__ __launch_bounds__(256) void prep_wt(const float* __restrict__ Wp,
                                               __bf16* __restrict__ WT) {
    int n = blockIdx.x, d = threadIdx.x;
    WT[n * C_ + d] = (__bf16)Wp[d * C_ + n];
}

// cproj[b][k][d] = bf16( sum_c cent[b][k][c] * Wc[c][d] + bc[d] )
__global__ __launch_bounds__(256) void prep_cproj(const float* __restrict__ cent,
                                                  const float* __restrict__ Wc,
                                                  const float* __restrict__ bc,
                                                  __bf16* __restrict__ cproj) {
    int bk = blockIdx.x;            // b*NC + k
    int d  = threadIdx.x;
    __shared__ float row[C_];
    row[d] = cent[(size_t)bk * C_ + d];
    __syncthreads();
    float acc = bc[d];
#pragma unroll 4
    for (int c = 0; c < C_; ++c) acc = fmaf(row[c], Wc[c * C_ + d], acc);
    cproj[(size_t)bk * C_ + d] = (__bf16)acc;
}

// CvT[b][c][k] = bf16(cent[b][k][c])   (B x 256 x 64)
__global__ __launch_bounds__(256) void prep_cvt(const float* __restrict__ cent,
                                                __bf16* __restrict__ CvT) {
    int b = blockIdx.x, c = threadIdx.x;
#pragma unroll 4
    for (int k = 0; k < NC_; ++k)
        CvT[((size_t)b * C_ + c) * NC_ + k] = (__bf16)cent[((size_t)b * NC_ + k) * C_ + c];
}

// ---------------- fused attention kernel ------------------------------------

__global__ __launch_bounds__(256) void attn_main(const float* __restrict__ pf,
                                                 const float* __restrict__ bp,
                                                 const __bf16* __restrict__ WT,
                                                 const __bf16* __restrict__ Cp,
                                                 const __bf16* __restrict__ CvT,
                                                 float* __restrict__ out) {
    __shared__ __attribute__((aligned(32))) unsigned char smem[65536];
    __bf16* P   = (__bf16*)smem;             // [128][256] bf16 (stage 1 output)
    float*  Ssh = (float*)smem;              // [128][64] f32  (reuses P after barrier)
    __bf16* Wt  = (__bf16*)(smem + 32768);   // [128][64] bf16 (softmax weights)

    const int nb   = N_ / MTILE;
    const int b    = blockIdx.x / nb;
    const int row0 = (blockIdx.x % nb) * MTILE;
    const int wave = threadIdx.x >> 5;
    const int lane = threadIdx.x & 31;
    const int l15  = lane & 15;
    const int lhi  = lane >> 4;
    const int lrow = wave * 16 + l15;        // local A-tile row for this lane

    // ---- load point rows once, convert to bf16 A fragments (registers) ----
    const float* prow = pf + ((size_t)b * N_ + row0 + lrow) * C_;
    v16bf Af[8];
#pragma unroll
    for (int ks = 0; ks < 8; ++ks) {
        const float* p0 = prow + ks * 32 + lhi * 8;
        float4 f0 = *(const float4*)(p0);
        float4 f1 = *(const float4*)(p0 + 4);
        float4 f2 = *(const float4*)(p0 + 16);
        float4 f3 = *(const float4*)(p0 + 20);
        v16bf a;
        a[0]=(__bf16)f0.x; a[1]=(__bf16)f0.y; a[2]=(__bf16)f0.z; a[3]=(__bf16)f0.w;
        a[4]=(__bf16)f1.x; a[5]=(__bf16)f1.y; a[6]=(__bf16)f1.z; a[7]=(__bf16)f1.w;
        a[8]=(__bf16)f2.x; a[9]=(__bf16)f2.y; a[10]=(__bf16)f2.z; a[11]=(__bf16)f2.w;
        a[12]=(__bf16)f3.x; a[13]=(__bf16)f3.y; a[14]=(__bf16)f3.z; a[15]=(__bf16)f3.w;
        Af[ks] = a;
    }

    // ---- Stage 1: P = pf @ W_p + b_p  (bf16 WMMA, f32 acc) ----
#pragma unroll 1
    for (int nt = 0; nt < 16; ++nt) {
        const int n0 = nt * 16;
        const float bias = bp[n0 + l15];
        v8f acc;
#pragma unroll
        for (int j = 0; j < 8; ++j) acc[j] = bias;
#pragma unroll
        for (int ks = 0; ks < 8; ++ks) {
            const __bf16* bq = WT + (size_t)(n0 + l15) * C_ + ks * 32 + lhi * 8;
            v16bf Bf = combine8(*(const v8bf*)bq, *(const v8bf*)(bq + 16));
            acc = __builtin_amdgcn_wmma_f32_16x16x32_bf16(false, Af[ks], false, Bf,
                                                          (short)0, acc, false, false);
        }
        __bf16* pdst = P + (size_t)(wave * 16 + lhi * 8) * C_ + n0 + l15;
#pragma unroll
        for (int j = 0; j < 8; ++j) pdst[(size_t)j * C_] = (__bf16)acc[j];
    }
    __syncthreads();

    // ---- cache P fragments for stage 2 (then P region is reused) ----
    v16bf Pf[8];
#pragma unroll
    for (int ks = 0; ks < 8; ++ks) {
        const __bf16* p0 = P + (size_t)lrow * C_ + ks * 32 + lhi * 8;
        v8bf lo = *(const v8bf*)p0;
        v8bf hi = *(const v8bf*)(p0 + 16);
        Pf[ks] = combine8(lo, hi);
    }
    __syncthreads();   // P dead -> its LDS becomes Ssh

    // ---- Stage 2: S = P @ c_proj^T ----
    const __bf16* Cpb = Cp + (size_t)b * NC_ * C_;
    v8f sacc[4];
#pragma unroll
    for (int t = 0; t < 4; ++t)
#pragma unroll
        for (int j = 0; j < 8; ++j) sacc[t][j] = 0.0f;
#pragma unroll
    for (int ks = 0; ks < 8; ++ks) {
#pragma unroll
        for (int t = 0; t < 4; ++t) {
            const __bf16* bq = Cpb + (size_t)(t * 16 + l15) * C_ + ks * 32 + lhi * 8;
            v16bf Bf = combine8(*(const v8bf*)bq, *(const v8bf*)(bq + 16));
            sacc[t] = __builtin_amdgcn_wmma_f32_16x16x32_bf16(false, Pf[ks], false, Bf,
                                                              (short)0, sacc[t], false, false);
        }
    }
#pragma unroll
    for (int t = 0; t < 4; ++t) {
        float* sdst = Ssh + (size_t)(wave * 16 + lhi * 8) * NC_ + t * 16 + l15;
#pragma unroll
        for (int j = 0; j < 8; ++j) sdst[(size_t)j * NC_] = sacc[t][j];
    }
    __syncthreads();

    // ---- softmax over the 64 centroids, one thread per row ----
    if (threadIdx.x < MTILE) {
        const float* srow = Ssh + (size_t)threadIdx.x * NC_;
        float m = srow[0];
#pragma unroll 4
        for (int k = 1; k < NC_; ++k) m = fmaxf(m, srow[k]);
        float s = 0.0f;
#pragma unroll 4
        for (int k = 0; k < NC_; ++k) s += __expf(srow[k] - m);
        const float inv = __frcp_rn(s);
        __bf16* wrow = Wt + (size_t)threadIdx.x * NC_;
#pragma unroll 4
        for (int k = 0; k < NC_; ++k) wrow[k] = (__bf16)(__expf(srow[k] - m) * inv);
    }
    __syncthreads();

    // ---- Stage 3: O = softmax @ V  + residual ----
    v16bf Wf[2];
#pragma unroll
    for (int ks = 0; ks < 2; ++ks) {
        const __bf16* p0 = Wt + (size_t)lrow * NC_ + ks * 32 + lhi * 8;
        v8bf lo = *(const v8bf*)p0;
        v8bf hi = *(const v8bf*)(p0 + 16);
        Wf[ks] = combine8(lo, hi);
    }
    const __bf16* CvTb = CvT + (size_t)b * C_ * NC_;
    const size_t gbase = ((size_t)b * N_ + row0 + wave * 16 + lhi * 8) * C_;
#pragma unroll 1
    for (int nt = 0; nt < 16; ++nt) {
        const int n0 = nt * 16;
        v8f oacc;
#pragma unroll
        for (int j = 0; j < 8; ++j) oacc[j] = 0.0f;
#pragma unroll
        for (int ks = 0; ks < 2; ++ks) {
            const __bf16* bq = CvTb + (size_t)(n0 + l15) * NC_ + ks * 32 + lhi * 8;
            v16bf Bf = combine8(*(const v8bf*)bq, *(const v8bf*)(bq + 16));
            oacc = __builtin_amdgcn_wmma_f32_16x16x32_bf16(false, Wf[ks], false, Bf,
                                                           (short)0, oacc, false, false);
        }
#pragma unroll
        for (int j = 0; j < 8; ++j) {
            size_t idx = gbase + (size_t)j * C_ + n0 + l15;
            out[idx] = pf[idx] + oacc[j];
        }
    }
}

// ---------------- launch ----------------------------------------------------

extern "C" void kernel_launch(void* const* d_in, const int* in_sizes, int n_in,
                              void* d_out, int out_size, void* d_ws, size_t ws_size,
                              hipStream_t stream) {
    const float* pf = (const float*)d_in[0];   // (B,N,C)
    const float* cf = (const float*)d_in[1];   // (B,NC,C)
    const float* Wp = (const float*)d_in[2];   // (C,C)
    const float* bp = (const float*)d_in[3];   // (C,)
    const float* Wc = (const float*)d_in[4];   // (C,C)
    const float* bc = (const float*)d_in[5];   // (C,)
    float* out = (float*)d_out;

    char* ws = (char*)d_ws;
    __bf16* WT  = (__bf16*)(ws);            // 256*256*2      = 131072 B
    __bf16* Cp  = (__bf16*)(ws + 131072);   // 8*64*256*2     = 262144 B
    __bf16* CvT = (__bf16*)(ws + 393216);   // 8*256*64*2     = 262144 B

    prep_wt   <<<C_,        256, 0, stream>>>(Wp, WT);
    prep_cproj<<<B_ * NC_,  256, 0, stream>>>(cf, Wc, bc, Cp);
    prep_cvt  <<<B_,        256, 0, stream>>>(cf, CvT);
    attn_main <<<B_ * (N_ / MTILE), 256, 0, stream>>>(pf, bp, WT, Cp, CvT, out);
}
